// Decoder_29875792511053
// MI455X (gfx1250) — compile-verified, matrix operand order checked by reading, not run on hardware
//
#include <hip/hip_runtime.h>
#include <hip/hip_bf16.h>
#include <math.h>

// ---------------- problem constants ----------------
constexpr int Bc   = 64;
constexpr int Tc   = 32;
constexpr int Ec   = 512;
constexpr int Hc   = 1024;
constexpr int VOCc = 10000;
constexpr int Rc   = 49;
constexpr int TWOE  = 2 * Ec;   // 1024
constexpr int FOURH = 4 * Hc;   // 4096
constexpr int BT    = Bc * Tc;  // 2048
constexpr int KCAT  = 2048;     // [c_hat | hid] concat width
constexpr int ATTLD = 64;       // padded ld for R=49 attention outputs

typedef __attribute__((ext_vector_type(16))) __bf16 v16bf;
typedef __attribute__((ext_vector_type(8)))  __bf16 v8bf;
typedef __attribute__((ext_vector_type(8)))  float  v8f;

__device__ __forceinline__ float sigf(float x) { return 1.0f / (1.0f + expf(-x)); }

// ============================================================
// WMMA bf16 GEMM:  C[M,N] (f32) = A[M,K] * Bw[N,K]^T (+C)(+bias)
// A, Bw row-major bf16, K % 32 == 0.
// 128x128 block tile, 256 threads (8 waves as 4x2), each wave owns
// 32(M) x 64(N): 2 A-frags x 4 B-frags -> 8 WMMA per K-step.
// Double-buffered LDS; row clamping instead of predicated loads
// (garbage rows/cols are never stored by the bounds-checked epilogue).
// ============================================================
constexpr int BM = 128, BN = 128, BKs = 32, LDSM = 48; // 96B LDS row (16B aligned)

__global__ __launch_bounds__(256)
void k_gemm_bf16(const __bf16* __restrict__ A, int lda,
                 const __bf16* __restrict__ Bw, int ldb,
                 float* __restrict__ C, int ldc,
                 const float* __restrict__ bias,
                 int M, int N, int K, int accumulate)
{
    __shared__ __bf16 Asm[2][BM * LDSM];
    __shared__ __bf16 Bsm[2][BN * LDSM];

    const int tid  = threadIdx.x;
    const int wave = tid >> 5;
    const int lane = tid & 31;
    const int hh   = lane >> 4;   // half-wave select
    const int lm   = lane & 15;
    const int wm   = wave >> 1;   // 0..3  (M direction, 32 rows each)
    const int wn   = wave & 1;    // 0..1  (N direction, 64 cols each)
    const int blockM = blockIdx.x * BM;
    const int blockN = blockIdx.y * BN;

    // ---- per-thread tile-load coordinates: 2 chunks A + 2 chunks B ----
    const int lr = tid >> 2;          // 0..63
    const int lc = (tid & 3) * 8;     // 0,8,16,24
    const size_t aoff0 = (size_t)min(blockM + lr,      M - 1) * lda + lc;
    const size_t aoff1 = (size_t)min(blockM + lr + 64, M - 1) * lda + lc;
    const size_t boff0 = (size_t)min(blockN + lr,      N - 1) * ldb + lc;
    const size_t boff1 = (size_t)min(blockN + lr + 64, N - 1) * ldb + lc;

    // ---- preload K-tile 0 ----
    v8bf a0 = *(const v8bf*)(A  + aoff0);
    v8bf a1 = *(const v8bf*)(A  + aoff1);
    v8bf b0 = *(const v8bf*)(Bw + boff0);
    v8bf b1 = *(const v8bf*)(Bw + boff1);
    *(v8bf*)(&Asm[0][(size_t)lr * LDSM + lc])        = a0;
    *(v8bf*)(&Asm[0][(size_t)(lr + 64) * LDSM + lc]) = a1;
    *(v8bf*)(&Bsm[0][(size_t)lr * LDSM + lc])        = b0;
    *(v8bf*)(&Bsm[0][(size_t)(lr + 64) * LDSM + lc]) = b1;
    __syncthreads();

    v8f acc[2][4] = {};
    const int nsteps = K / BKs;

    for (int s = 0; s < nsteps; ++s) {
        const int cur = s & 1;
        const int nxt = cur ^ 1;

        // issue global loads for next K-tile early (hide HBM/L2 latency)
        if (s + 1 < nsteps) {
            const int k1 = (s + 1) * BKs;
            a0 = *(const v8bf*)(A  + aoff0 + k1);
            a1 = *(const v8bf*)(A  + aoff1 + k1);
            b0 = *(const v8bf*)(Bw + boff0 + k1);
            b1 = *(const v8bf*)(Bw + boff1 + k1);
        }
        // prefetch K-tile s+2 into cache (global_prefetch_b8)
        if (s + 2 < nsteps) {
            const int k2 = (s + 2) * BKs;
            __builtin_prefetch((const char*)(A  + aoff0 + k2), 0, 1);
            __builtin_prefetch((const char*)(A  + aoff1 + k2), 0, 1);
            __builtin_prefetch((const char*)(Bw + boff0 + k2), 0, 1);
            __builtin_prefetch((const char*)(Bw + boff1 + k2), 0, 1);
        }

        // ---- fragments from LDS[cur]; K(e) = e%8 + 8*hh + 16*(e/8) ----
        v16bf af[2];
        #pragma unroll
        for (int mi = 0; mi < 2; ++mi) {
            const int arow = wm * 32 + mi * 16 + lm;
            v8bf x0 = *(const v8bf*)(&Asm[cur][(size_t)arow * LDSM + 8 * hh]);
            v8bf x1 = *(const v8bf*)(&Asm[cur][(size_t)arow * LDSM + 16 + 8 * hh]);
            af[mi] = __builtin_shufflevector(x0, x1, 0,1,2,3,4,5,6,7,8,9,10,11,12,13,14,15);
        }
        #pragma unroll
        for (int ni = 0; ni < 4; ++ni) {
            const int brow = wn * 64 + ni * 16 + lm;
            v8bf y0 = *(const v8bf*)(&Bsm[cur][(size_t)brow * LDSM + 8 * hh]);
            v8bf y1 = *(const v8bf*)(&Bsm[cur][(size_t)brow * LDSM + 16 + 8 * hh]);
            v16bf bfrag = __builtin_shufflevector(y0, y1, 0,1,2,3,4,5,6,7,8,9,10,11,12,13,14,15);
            #pragma unroll
            for (int mi = 0; mi < 2; ++mi) {
                acc[mi][ni] = __builtin_amdgcn_wmma_f32_16x16x32_bf16(
                    false, af[mi], false, bfrag, (short)0, acc[mi][ni], false, false);
            }
        }

        // ---- stage next K-tile into the other LDS buffer ----
        if (s + 1 < nsteps) {
            *(v8bf*)(&Asm[nxt][(size_t)lr * LDSM + lc])        = a0;
            *(v8bf*)(&Asm[nxt][(size_t)(lr + 64) * LDSM + lc]) = a1;
            *(v8bf*)(&Bsm[nxt][(size_t)lr * LDSM + lc])        = b0;
            *(v8bf*)(&Bsm[nxt][(size_t)(lr + 64) * LDSM + lc]) = b1;
        }
        __syncthreads();
    }

    // ---- epilogue: C/D layout row = vgpr + 8*hh, col = lm ----
    #pragma unroll
    for (int ni = 0; ni < 4; ++ni) {
        const int col = blockN + wn * 64 + ni * 16 + lm;
        if (col >= N) continue;
        #pragma unroll
        for (int mi = 0; mi < 2; ++mi) {
            #pragma unroll
            for (int i = 0; i < 8; ++i) {
                const int row = blockM + wm * 32 + mi * 16 + hh * 8 + i;
                if (row >= M) continue;
                size_t idx = (size_t)row * ldc + col;
                float v = acc[mi][ni][i];
                if (accumulate) v += C[idx];
                if (bias)       v += bias[col];
                C[idx] = v;
            }
        }
    }
}

// ============================================================
// Pointwise / setup kernels
// ============================================================
__global__ void k_zero_u32(unsigned* p, size_t n) {
    size_t i = (size_t)blockIdx.x * blockDim.x + threadIdx.x;
    size_t st = (size_t)gridDim.x * blockDim.x;
    for (; i < n; i += st) p[i] = 0u;
}

__global__ void k_f32_to_bf16(const float* __restrict__ s, __bf16* __restrict__ d, size_t n) {
    size_t i = (size_t)blockIdx.x * blockDim.x + threadIdx.x;
    size_t st = (size_t)gridDim.x * blockDim.x;
    for (; i < n; i += st) d[i] = (__bf16)s[i];
}

// x[bt] = [ embed_W[caps[bt]] , v_g[b] ]  (bf16, ld = 2E)
__global__ void k_build_x(const int* __restrict__ caps, const float* __restrict__ embW,
                          const float* __restrict__ vg, __bf16* __restrict__ xb)
{
    int row = blockIdx.x;            // b*T + t
    int b = row / Tc;
    int tok = caps[row];
    for (int e = threadIdx.x; e < Ec; e += blockDim.x) {
        xb[(size_t)row * TWOE + e]      = (__bf16)embW[(size_t)tok * Ec + e];
        xb[(size_t)row * TWOE + Ec + e] = (__bf16)vg[(size_t)b * Ec + e];
    }
}

// Wcat[n][k] = k<H ? mlp_W[n][k] : h2score_W[n][k-H]
__global__ void k_build_wcat(const float* __restrict__ w1, const float* __restrict__ w2,
                             __bf16* __restrict__ wcat)
{
    size_t total = (size_t)VOCc * KCAT;
    size_t i = (size_t)blockIdx.x * blockDim.x + threadIdx.x;
    size_t st = (size_t)gridDim.x * blockDim.x;
    for (; i < total; i += st) {
        size_t n = i / KCAT, k = i % KCAT;
        float v = (k < (size_t)Hc) ? w1[n * Hc + k] : w2[n * Hc + (k - Hc)];
        wcat[i] = (__bf16)v;
    }
}

__global__ void k_build_bias(const float* a, const float* b, float* o) {
    int i = blockIdx.x * blockDim.x + threadIdx.x;
    if (i < VOCc) o[i] = a[i] + b[i];
}

// LSTM gate update for timestep t.
__global__ void k_lstm_gate(const float* __restrict__ gbuf,   // [B,4H] h@Whh^T
                            const float* __restrict__ xW,     // [B*T,4H] x@Wih^T
                            const float* __restrict__ bih, const float* __restrict__ bhh,
                            float* __restrict__ cbuf,         // [B,H] in/out
                            __bf16* __restrict__ hbf,         // [B,H] next-step A operand
                            __bf16* __restrict__ hidbf,       // [B*T,H]
                            __bf16* __restrict__ hidprevbf,   // [B*T,H] shifted
                            __bf16* __restrict__ ccat,        // [B*T,2048], cols H..2H-1
                            float* __restrict__ memf,         // [B*T,H]
                            float* __restrict__ outHT, float* __restrict__ outCT, int t)
{
    int idx = blockIdx.x * blockDim.x + threadIdx.x;
    if (idx >= Bc * Hc) return;
    int b = idx / Hc, j = idx % Hc;
    size_t g4 = (size_t)b * FOURH;
    size_t x4 = (size_t)(b * Tc + t) * FOURH;
    float gi = gbuf[g4 + j]          + xW[x4 + j]          + bih[j]          + bhh[j];
    float gf = gbuf[g4 + Hc + j]     + xW[x4 + Hc + j]     + bih[Hc + j]     + bhh[Hc + j];
    float gg = gbuf[g4 + 2*Hc + j]   + xW[x4 + 2*Hc + j]   + bih[2*Hc + j]   + bhh[2*Hc + j];
    float go = gbuf[g4 + 3*Hc + j]   + xW[x4 + 3*Hc + j]   + bih[3*Hc + j]   + bhh[3*Hc + j];
    float c = sigf(gf) * cbuf[idx] + sigf(gi) * tanhf(gg);
    float h = sigf(go) * tanhf(c);
    cbuf[idx] = c;
    hbf[idx] = (__bf16)h;
    size_t hrow = (size_t)(b * Tc + t);
    hidbf[hrow * Hc + j] = (__bf16)h;
    ccat[hrow * KCAT + Hc + j] = (__bf16)h;
    if (t + 1 < Tc) hidprevbf[(hrow + 1) * Hc + j] = (__bf16)h;
    memf[hrow * Hc + j] = c;
    if (t == Tc - 1) { outHT[idx] = h; outCT[idx] = c; }
}

// s_t = sigmoid(gatepre) * tanh(mem)
__global__ void k_st(const float* __restrict__ gp, const float* __restrict__ memf,
                     float* __restrict__ stf, __bf16* __restrict__ stb)
{
    size_t n = (size_t)BT * Hc;
    size_t i = (size_t)blockIdx.x * blockDim.x + threadIdx.x;
    size_t st = (size_t)gridDim.x * blockDim.x;
    for (; i < n; i += st) {
        float s = sigf(gp[i]) * tanhf(memf[i]);
        stf[i] = s;
        stb[i] = (__bf16)s;
    }
}

// z_v[bt,r] = sum_j attW[j] * tanh(Vatt[b*R+r, j] + hAtt[bt, j])
__global__ void k_zv(const float* __restrict__ Vatt, const float* __restrict__ hAtt,
                     const float* __restrict__ attW, float* __restrict__ zv)
{
    int i = blockIdx.x * blockDim.x + threadIdx.x;
    if (i >= BT * Rc) return;
    int r = i % Rc, bt = i / Rc, b = bt / Tc;
    const float* va = Vatt + (size_t)(b * Rc + r) * ATTLD;
    const float* ha = hAtt + (size_t)bt * ATTLD;
    float s = 0.0f;
    for (int j = 0; j < Rc; ++j) s += attW[j] * tanhf(va[j] + ha[j]);
    zv[(size_t)bt * Rc + r] = s;
}

// z_s[bt] = sum_j attW[j] * tanh(sAtt[bt,j] + hAtt2[bt,j])
__global__ void k_zs(const float* __restrict__ sAtt, const float* __restrict__ hAtt2,
                     const float* __restrict__ attW, float* __restrict__ zs)
{
    int bt = blockIdx.x * blockDim.x + threadIdx.x;
    if (bt >= BT) return;
    const float* sa = sAtt  + (size_t)bt * ATTLD;
    const float* ha = hAtt2 + (size_t)bt * ATTLD;
    float s = 0.0f;
    for (int j = 0; j < Rc; ++j) s += attW[j] * tanhf(sa[j] + ha[j]);
    zs[bt] = s;
}

// softmax over [z_v(49), z_s]; emit alpha (ws), alpha*(1-beta) and beta (d_out)
__global__ void k_softmax(const float* __restrict__ zv, const float* __restrict__ zs,
                          float* __restrict__ alpha, float* __restrict__ outAlpha,
                          float* __restrict__ outBeta)
{
    int bt = blockIdx.x * blockDim.x + threadIdx.x;
    if (bt >= BT) return;
    float vals[Rc + 1];
    float mx = -1e30f;
    for (int r = 0; r < Rc; ++r) { vals[r] = zv[(size_t)bt * Rc + r]; mx = fmaxf(mx, vals[r]); }
    vals[Rc] = zs[bt]; mx = fmaxf(mx, vals[Rc]);
    float sum = 0.0f;
    for (int r = 0; r <= Rc; ++r) { vals[r] = expf(vals[r] - mx); sum += vals[r]; }
    float inv = 1.0f / sum;
    float beta = vals[Rc] * inv;
    outBeta[bt] = beta;
    float omb = 1.0f - beta;
    for (int r = 0; r < Rc; ++r) {
        float a = vals[r] * inv;
        alpha[(size_t)bt * Rc + r]    = a;
        outAlpha[(size_t)bt * Rc + r] = a * omb;
    }
}

// c_hat[bt,h] = beta*s_t + sum_r alpha[r]*V[b,r,h]  -> ccat cols 0..H-1 (bf16)
__global__ void k_chat(const float* __restrict__ alpha, const float* __restrict__ beta,
                       const float* __restrict__ stf, const float* __restrict__ Vf,
                       __bf16* __restrict__ ccat)
{
    int bt = blockIdx.x, b = bt / Tc;
    __shared__ float al[Rc];
    __shared__ float be;
    if (threadIdx.x < Rc) al[threadIdx.x] = alpha[(size_t)bt * Rc + threadIdx.x];
    if (threadIdx.x == 0) be = beta[bt];
    __syncthreads();
    for (int h = threadIdx.x; h < Hc; h += blockDim.x) {
        float s = be * stf[(size_t)bt * Hc + h];
        for (int r = 0; r < Rc; ++r) s += al[r] * Vf[((size_t)b * Rc + r) * Hc + h];
        ccat[(size_t)bt * KCAT + h] = (__bf16)s;
    }
}

// ============================================================
extern "C" void kernel_launch(void* const* d_in, const int* in_sizes, int n_in,
                              void* d_out, int out_size, void* d_ws, size_t ws_size,
                              hipStream_t stream)
{
    const float* V     = (const float*)d_in[0];
    const float* vg    = (const float*)d_in[1];
    const int*   caps  = (const int*)d_in[2];
    const float* embW  = (const float*)d_in[3];
    const float* Wih   = (const float*)d_in[4];
    const float* Whh   = (const float*)d_in[5];
    const float* bih   = (const float*)d_in[6];
    const float* bhh   = (const float*)d_in[7];
    const float* sh2s  = (const float*)d_in[8];
    const float* sx2s  = (const float*)d_in[9];
    const float* ah2   = (const float*)d_in[10];
    const float* ah22  = (const float*)d_in[11];
    const float* av2   = (const float*)d_in[12];
    const float* as2   = (const float*)d_in[13];
    const float* attW  = (const float*)d_in[14];
    const float* mlpW  = (const float*)d_in[15];
    const float* mlpb  = (const float*)d_in[16];
    const float* h2sW  = (const float*)d_in[17];
    const float* h2sb  = (const float*)d_in[18];

    float* out       = (float*)d_out;
    float* outScores = out;                                   // B*T*VOC
    float* outHT     = outScores + (size_t)BT * VOCc;         // B*H
    float* outCT     = outHT + (size_t)Bc * Hc;               // B*H
    float* outAlpha  = outCT + (size_t)Bc * Hc;               // B*T*R
    float* outBeta   = outAlpha + (size_t)BT * Rc;            // B*T

    // ---- workspace carve ----
    char* wsb = (char*)d_ws;
    size_t off = 0;
    auto carve = [&](size_t bytes) -> void* {
        void* p = wsb + off;
        off += (bytes + 255) & ~(size_t)255;
        return p;
    };
    __bf16* xb        = (__bf16*)carve((size_t)BT * TWOE * 2);
    __bf16* Wihb      = (__bf16*)carve((size_t)FOURH * TWOE * 2);
    __bf16* Whhb      = (__bf16*)carve((size_t)FOURH * Hc * 2);
    float*  xWih      = (float*)carve((size_t)BT * FOURH * 4);
    float*  gbuf      = (float*)carve((size_t)Bc * FOURH * 4);
    __bf16* hbf       = (__bf16*)carve((size_t)Bc * Hc * 2);
    float*  cbuf      = (float*)carve((size_t)Bc * Hc * 4);
    __bf16* hidbf     = (__bf16*)carve((size_t)BT * Hc * 2);
    __bf16* hidprevbf = (__bf16*)carve((size_t)BT * Hc * 2);
    float*  memf      = (float*)carve((size_t)BT * Hc * 4);
    __bf16* sxsb      = (__bf16*)carve((size_t)Hc * TWOE * 2);
    __bf16* shsb      = (__bf16*)carve((size_t)Hc * TWOE * 2);
    float*  gatepre   = (float*)carve((size_t)BT * Hc * 4);
    float*  stf       = (float*)carve((size_t)BT * Hc * 4);
    __bf16* stb       = (__bf16*)carve((size_t)BT * Hc * 2);
    __bf16* Vb        = (__bf16*)carve((size_t)Bc * Rc * Hc * 2);
    __bf16* av2b      = (__bf16*)carve((size_t)Rc * Hc * 2);
    __bf16* ah2b      = (__bf16*)carve((size_t)Rc * Hc * 2);
    __bf16* ah22b     = (__bf16*)carve((size_t)Rc * Hc * 2);
    __bf16* as2b      = (__bf16*)carve((size_t)Rc * Hc * 2);
    float*  Vatt      = (float*)carve((size_t)Bc * Rc * ATTLD * 4);
    float*  hAtt      = (float*)carve((size_t)BT * ATTLD * 4);
    float*  hAtt2     = (float*)carve((size_t)BT * ATTLD * 4);
    float*  sAtt      = (float*)carve((size_t)BT * ATTLD * 4);
    float*  zv        = (float*)carve((size_t)BT * Rc * 4);
    float*  zs        = (float*)carve((size_t)BT * 4);
    float*  alpha     = (float*)carve((size_t)BT * Rc * 4);
    __bf16* ccat      = (__bf16*)carve((size_t)BT * KCAT * 2);
    __bf16* Wcatb     = (__bf16*)carve((size_t)VOCc * KCAT * 2);
    float*  biascat   = (float*)carve((size_t)VOCc * 4);

    auto cvt = [&](const float* s, __bf16* d, size_t n) {
        unsigned g = (unsigned)((n + 256 * 16 - 1) / (256 * 16));
        k_f32_to_bf16<<<g, 256, 0, stream>>>(s, d, n);
    };
    auto zero = [&](void* p, size_t bytes) {
        size_t n = bytes / 4;
        unsigned g = (unsigned)((n + 256 * 16 - 1) / (256 * 16));
        k_zero_u32<<<g, 256, 0, stream>>>((unsigned*)p, n);
    };
    auto gemm = [&](const __bf16* A, int lda, const __bf16* Bw, int ldb,
                    float* C, int ldc, const float* bias, int M, int N, int K, int acc) {
        dim3 g((M + BM - 1) / BM, (N + BN - 1) / BN);
        k_gemm_bf16<<<g, 256, 0, stream>>>(A, lda, Bw, ldb, C, ldc, bias, M, N, K, acc);
    };

    // ---- setup: conversions, concat builds, zero init ----
    cvt(Wih, Wihb, (size_t)FOURH * TWOE);
    cvt(Whh, Whhb, (size_t)FOURH * Hc);
    cvt(sx2s, sxsb, (size_t)Hc * TWOE);
    cvt(sh2s, shsb, (size_t)Hc * TWOE);
    cvt(av2, av2b, (size_t)Rc * Hc);
    cvt(ah2, ah2b, (size_t)Rc * Hc);
    cvt(ah22, ah22b, (size_t)Rc * Hc);
    cvt(as2, as2b, (size_t)Rc * Hc);
    cvt(V, Vb, (size_t)Bc * Rc * Hc);
    k_build_x<<<BT, 256, 0, stream>>>(caps, embW, vg, xb);
    k_build_wcat<<<4096, 256, 0, stream>>>(mlpW, h2sW, Wcatb);
    k_build_bias<<<(VOCc + 255) / 256, 256, 0, stream>>>(mlpb, h2sb, biascat);
    zero(hbf, (size_t)Bc * Hc * 2);
    zero(cbuf, (size_t)Bc * Hc * 4);
    zero(hidprevbf, (size_t)BT * Hc * 2);

    // ---- xWih = x @ Wih^T (one big GEMM for all timesteps) ----
    gemm(xb, TWOE, Wihb, TWOE, xWih, FOURH, nullptr, BT, FOURH, TWOE, 0);

    // ---- sequential LSTM: per-step h @ Whh^T + gate pointwise ----
    for (int t = 0; t < Tc; ++t) {
        gemm(hbf, Hc, Whhb, Hc, gbuf, FOURH, nullptr, Bc, FOURH, Hc, 0);
        k_lstm_gate<<<(Bc * Hc + 255) / 256, 256, 0, stream>>>(
            gbuf, xWih, bih, bhh, cbuf, hbf, hidbf, hidprevbf, ccat, memf,
            outHT, outCT, t);
    }

    // ---- sentinel gate: gatepre = x@s_x2s^T + hid_prev@s_h2s^T ; s_t ----
    gemm(xb, TWOE, sxsb, TWOE, gatepre, Hc, nullptr, BT, Hc, TWOE, 0);
    gemm(hidprevbf, Hc, shsb, TWOE, gatepre, Hc, nullptr, BT, Hc, Hc, 1);
    k_st<<<2048, 256, 0, stream>>>(gatepre, memf, stf, stb);

    // ---- attention projections ----
    gemm(Vb, Hc, av2b, Hc, Vatt, ATTLD, nullptr, Bc * Rc, Rc, Hc, 0);
    gemm(hidbf, Hc, ah2b, Hc, hAtt, ATTLD, nullptr, BT, Rc, Hc, 0);
    gemm(hidbf, Hc, ah22b, Hc, hAtt2, ATTLD, nullptr, BT, Rc, Hc, 0);
    gemm(stb, Hc, as2b, Hc, sAtt, ATTLD, nullptr, BT, Rc, Hc, 0);

    // ---- scores z, softmax, c_hat ----
    k_zv<<<(BT * Rc + 255) / 256, 256, 0, stream>>>(Vatt, hAtt, attW, zv);
    k_zs<<<(BT + 255) / 256, 256, 0, stream>>>(sAtt, hAtt2, attW, zs);
    k_softmax<<<(BT + 255) / 256, 256, 0, stream>>>(zv, zs, alpha, outAlpha, outBeta);
    k_chat<<<BT, 256, 0, stream>>>(alpha, outBeta, stf, V, ccat);

    // ---- final vocab projection: [c_hat|hid] @ [mlp_W|h2score_W]^T + biases ----
    gemm(ccat, KCAT, Wcatb, KCAT, outScores, VOCc, biascat, BT, VOCc, KCAT, 0);
}
